// MambaS2ST_13657996001383
// MI455X (gfx1250) — compile-verified, matrix operand order checked by reading, not run on hardware
//
#include <hip/hip_runtime.h>
#include <math.h>

#define D_MODEL  512
#define D_FFN    2048
#define N_LAYERS 6
#define D_STATE  16
#define D_CONV   4
#define D_INNER  1024
#define DT_RANK  32
#define BATCH    2
#define LT       512
#define LS       512

typedef __attribute__((ext_vector_type(16))) __bf16        v16bf;
typedef __attribute__((ext_vector_type(8)))  __bf16        v8bf;
typedef __attribute__((ext_vector_type(8)))  float          v8f;

// ---------------------------------------------------------------- helpers
__device__ __forceinline__ unsigned short f2bf_bits(float f) {
  unsigned int u = __builtin_bit_cast(unsigned int, f);
  u += 0x7FFFu + ((u >> 16) & 1u);           // round-to-nearest-even
  return (unsigned short)(u >> 16);
}
__device__ __forceinline__ __bf16 f2bf(float f) {
  return __builtin_bit_cast(__bf16, f2bf_bits(f));
}

// A/B fragment per ISA 7.12.2: lane holds K = base..base+7 (v0..v3) and
// K = base+16..base+23 (v4..v7), base = (lane/16)*8. Two aligned b128 loads.
__device__ __forceinline__ v16bf frag(const __bf16* __restrict__ p) {
  v8bf lo = *(const v8bf*)(p);
  v8bf hi = *(const v8bf*)(p + 16);
  return __builtin_shufflevector(lo, hi, 0, 1, 2, 3, 4, 5, 6, 7,
                                 8, 9, 10, 11, 12, 13, 14, 15);
}

// ------------------------------------------------ fp32 -> bf16 pack
__global__ __launch_bounds__(256)
void pack_bf16_kernel(const float* __restrict__ in, __bf16* __restrict__ out, int n)
{
  int i = blockIdx.x * 256 + threadIdx.x;
  if (i < n) out[i] = f2bf(in[i]);
}

// ------------------------------------------------ WMMA GEMM: C = act(A @ W^T + bias)
// A: (M,K) bf16 stride lda; W: (N,K) bf16 stride ldw; C fp32 / Cb bf16 (either may
// be null), stride ldc. act: 0=none, 1=relu, 2=softplus.
// Block = 8 waves -> 64(M) x 128(N); wave = 32x32 (4 WMMAs / k-step, 2x reuse).
__global__ __launch_bounds__(256)
void gemm_bf16_wmma(const __bf16* __restrict__ A, int lda,
                    const __bf16* __restrict__ W, int ldw,
                    const float* __restrict__ bias,
                    float* __restrict__ C, __bf16* __restrict__ Cb, int ldc,
                    int M, int N, int K, int act)
{
  const int lane = threadIdx.x & 31;
  const int wave = threadIdx.x >> 5;
  const int wm = wave >> 2, wn = wave & 3;
  const int m0 = blockIdx.y * 64 + wm * 32;
  const int n0 = blockIdx.x * 128 + wn * 32;
  if (m0 >= M || n0 >= N) return;

  const int half = lane >> 4;      // 0 or 1
  const int lr   = lane & 15;

  const __bf16* arow0 = A + (size_t)(m0 + lr) * lda;
  const __bf16* arow1 = A + (size_t)(m0 + 16 + lr) * lda;
  const __bf16* wrow0 = W + (size_t)(n0 + lr) * ldw;
  const __bf16* wrow1 = W + (size_t)(n0 + 16 + lr) * ldw;

  v8f c00 = {}, c01 = {}, c10 = {}, c11 = {};
  for (int k0 = 0; k0 < K; k0 += 32) {
    const int kb = k0 + half * 8;
    if (k0 + 32 < K) {
      __builtin_prefetch(arow0 + kb + 32, 0, 1);   // global_prefetch_b8
      __builtin_prefetch(arow1 + kb + 32, 0, 1);
    }
    v16bf a0 = frag(arow0 + kb);
    v16bf a1 = frag(arow1 + kb);
    v16bf b0 = frag(wrow0 + kb);
    v16bf b1 = frag(wrow1 + kb);
    c00 = __builtin_amdgcn_wmma_f32_16x16x32_bf16(false, a0, false, b0,
                                                  (short)0, c00, false, false);
    c01 = __builtin_amdgcn_wmma_f32_16x16x32_bf16(false, a0, false, b1,
                                                  (short)0, c01, false, false);
    c10 = __builtin_amdgcn_wmma_f32_16x16x32_bf16(false, a1, false, b0,
                                                  (short)0, c10, false, false);
    c11 = __builtin_amdgcn_wmma_f32_16x16x32_bf16(false, a1, false, b1,
                                                  (short)0, c11, false, false);
  }

  // C/D layout: lane -> N = lane%16; VGPR v -> M = v + (lane/16)*8
  const int cn[2] = { n0 + lr, n0 + 16 + lr };
  const float bv[2] = { bias ? bias[cn[0]] : 0.f, bias ? bias[cn[1]] : 0.f };
#pragma unroll
  for (int mi = 0; mi < 2; ++mi) {
#pragma unroll
    for (int ni = 0; ni < 2; ++ni) {
      v8f acc = (mi == 0) ? (ni == 0 ? c00 : c01) : (ni == 0 ? c10 : c11);
#pragma unroll
      for (int v = 0; v < 8; ++v) {
        const int m = m0 + mi * 16 + v + half * 8;
        float r = acc[v] + bv[ni];
        if (act == 1) r = fmaxf(r, 0.f);
        else if (act == 2) r = (r > 20.f) ? r : log1pf(__expf(r));
        const size_t o = (size_t)m * ldc + cn[ni];
        if (C)  C[o]  = r;
        if (Cb) Cb[o] = f2bf(r);
      }
    }
  }
}

// ------------------------------------------------ embedding + sinusoidal PE
__global__ __launch_bounds__(256)
void embed_kernel(const int* __restrict__ tgt, const float* __restrict__ emb,
                  float* __restrict__ x, __bf16* __restrict__ xb)
{
  int idx = blockIdx.x * 256 + threadIdx.x;           // over B*LT*D_MODEL
  if (idx >= BATCH * LT * D_MODEL) return;
  int k   = idx & (D_MODEL - 1);
  int row = idx >> 9;                                 // b*LT + t
  int t   = row & (LT - 1);
  float v = emb[(size_t)tgt[row] * D_MODEL + k] * 22.62741699796952f; // sqrt(512)
  float dv = __expf(-(float)(k & ~1) * (9.210340371976184f / (float)D_MODEL));
  float ang = (float)t * dv;
  v += (k & 1) ? __cosf(ang) : __sinf(ang);
  x[idx]  = v;
  xb[idx] = f2bf(v);
}

// ------------------------------------------------ cat = concat([src, x], time) (bf16)
__global__ __launch_bounds__(256)
void concat_kernel(const float* __restrict__ src, const float* __restrict__ x,
                   __bf16* __restrict__ catb)
{
  int idx = blockIdx.x * 256 + threadIdx.x;           // over B*1024*D_MODEL
  if (idx >= BATCH * (LS + LT) * D_MODEL) return;
  int k   = idx & (D_MODEL - 1);
  int row = idx >> 9;                                 // b*1024 + l
  int b = row >> 10, l = row & 1023;
  float v = (l < LS) ? src[((size_t)b * LS + l) * D_MODEL + k]
                     : x  [((size_t)b * LT + (l - LS)) * D_MODEL + k];
  catb[idx] = f2bf(v);
}

// ------------------------------------------------ causal depthwise conv + SiLU
__global__ __launch_bounds__(256)
void conv_silu_kernel(const float* __restrict__ xz, float* __restrict__ xc,
                      __bf16* __restrict__ xcb,
                      const float* __restrict__ cw, const float* __restrict__ cb,
                      int rows, int L)
{
  int idx = blockIdx.x * 256 + threadIdx.x;           // over rows*D_INNER
  if (idx >= rows * D_INNER) return;
  int c = idx & (D_INNER - 1);
  int r = idx >> 10;
  int t = r % L;
  const float* base = xz + (size_t)r * (2 * D_INNER) + c;
  float acc = cb[c];
#pragma unroll
  for (int j = 0; j < D_CONV; ++j) {
    int ts = t - (D_CONV - 1) + j;
    if (ts >= 0) acc += cw[c * D_CONV + j] * base[(j - (D_CONV - 1)) * (2 * D_INNER)];
  }
  float s = acc / (1.f + __expf(-acc));               // silu
  xc[idx]  = s;
  xcb[idx] = f2bf(s);
}

// ------------------------------------------------ selective scan (sequential in t)
// One thread per (b,d); h[16] in registers. B_t/C_t broadcast fetched with the
// CDNA5 async LDS DMA (global_load_async_to_lds_b128 + s_wait_asynccnt).
__global__ __launch_bounds__(256)
void scan_kernel(const float* __restrict__ dbl, const float* __restrict__ dt,
                 const float* __restrict__ xc, const float* __restrict__ z,
                 const float* __restrict__ Alog, const float* __restrict__ Dsk,
                 __bf16* __restrict__ yb, int L)
{
  const int gd = blockIdx.x * 256 + threadIdx.x;      // over BATCH*D_INNER
  const int b = gd >> 10, d = gd & (D_INNER - 1);
  float A[D_STATE], h[D_STATE];
#pragma unroll
  for (int n = 0; n < D_STATE; ++n) {
    A[n] = -__expf(Alog[(size_t)d * D_STATE + n]);
    h[n] = 0.f;
  }
  const float Dv = Dsk[d];
  __shared__ __align__(16) float sBC[2 * D_STATE];    // [B(16) | C(16)]
  const size_t rb = (size_t)b * L;
  for (int t = 0; t < L; ++t) {
    __syncthreads();                                  // protect sBC (WAR)
    if (threadIdx.x < 8) {
      unsigned long long ga = (unsigned long long)(uintptr_t)
          (dbl + (rb + t) * 64 + DT_RANK + threadIdx.x * 4);
      unsigned la = (unsigned)(uintptr_t)(&sBC[threadIdx.x * 4]);
      asm volatile("global_load_async_to_lds_b128 %0, %1, off"
                   :: "v"(la), "v"(ga) : "memory");
    }
    asm volatile("s_wait_asynccnt 0x0" ::: "memory");
    __syncthreads();
    const size_t ro = rb + t;
    const float dtv = dt[ro * D_INNER + d];
    const float u   = xc[ro * D_INNER + d];
    const float du  = dtv * u;
    float yv = 0.f;
#pragma unroll
    for (int n = 0; n < D_STATE; ++n) {
      h[n] = __expf(dtv * A[n]) * h[n] + du * sBC[n];
      yv += h[n] * sBC[D_STATE + n];
    }
    const float zv  = z[ro * (2 * D_INNER) + d];
    const float sil = zv / (1.f + __expf(-zv));
    yb[ro * D_INNER + d] = f2bf((yv + u * Dv) * sil);
  }
}

// ------------------------------------------------ out = LN(xin + delta) * g + b
__global__ __launch_bounds__(256)
void add_ln_kernel(const float* __restrict__ xin, const float* __restrict__ delta,
                   int dstride, int doff,
                   const float* __restrict__ g, const float* __restrict__ bta,
                   float* __restrict__ out, __bf16* __restrict__ outb)
{
  __shared__ float sred[256];
  const int row = blockIdx.x;                         // 0..B*LT-1
  const int bi = row / LT, t = row - bi * LT;
  const int tid = threadIdx.x;
  const float* xr = xin + (size_t)row * D_MODEL;
  float v0 = xr[tid], v1 = xr[tid + 256];
  if (delta) {
    const float* dr = delta + ((size_t)bi * dstride + doff + t) * D_MODEL;
    v0 += dr[tid]; v1 += dr[tid + 256];
  }
  sred[tid] = v0 + v1;
  __syncthreads();
  for (int st = 128; st > 0; st >>= 1) {
    if (tid < st) sred[tid] += sred[tid + st];
    __syncthreads();
  }
  const float mean = sred[0] * (1.f / D_MODEL);
  __syncthreads();
  const float d0 = v0 - mean, d1 = v1 - mean;
  sred[tid] = d0 * d0 + d1 * d1;
  __syncthreads();
  for (int st = 128; st > 0; st >>= 1) {
    if (tid < st) sred[tid] += sred[tid + st];
    __syncthreads();
  }
  const float rstd = rsqrtf(sred[0] * (1.f / D_MODEL) + 1e-6f);
  const float r0 = d0 * rstd * g[tid]       + bta[tid];
  const float r1 = d1 * rstd * g[tid + 256] + bta[tid + 256];
  float* orow = out + (size_t)row * D_MODEL;
  orow[tid] = r0; orow[tid + 256] = r1;
  if (outb) {
    __bf16* obr = outb + (size_t)row * D_MODEL;
    obr[tid] = f2bf(r0); obr[tid + 256] = f2bf(r1);
  }
}

// ================================================ host-side orchestration
static inline void launch_gemm(const __bf16* A, int lda, const __bf16* W, int ldw,
                               const float* bias, float* C, __bf16* Cb, int ldc,
                               int M, int N, int K, int act, hipStream_t s)
{
  dim3 grid((N + 127) / 128, (M + 63) / 64, 1);
  gemm_bf16_wmma<<<grid, 256, 0, s>>>(A, lda, W, ldw, bias, C, Cb, ldc, M, N, K, act);
}

static inline void launch_pack(const float* in, __bf16* out, size_t n, hipStream_t s)
{
  pack_bf16_kernel<<<((int)n + 255) / 256, 256, 0, s>>>(in, out, (int)n);
}

struct MambaW {                 // per-branch packed bf16 weights (all layers)
  __bf16 *in_w, *xproj, *dtw, *outw;
};

static void run_mamba(const __bf16* xinb, int L, int layer, const MambaW& wb,
                      const float* conv_w, const float* conv_b,
                      const float* dtb, const float* Alog, const float* Dsk,
                      float* xz, float* xc, __bf16* xcb, float* dbl, __bf16* dblb,
                      float* dt, __bf16* ysb, float* t2, hipStream_t s)
{
  const int M = BATCH * L;
  const __bf16* in_w  = wb.in_w  + (size_t)layer * 2 * D_INNER * D_MODEL;
  const __bf16* xproj = wb.xproj + (size_t)layer * 64 * D_INNER;
  const __bf16* dtw   = wb.dtw   + (size_t)layer * D_INNER * DT_RANK;
  const __bf16* outw  = wb.outw  + (size_t)layer * D_MODEL * D_INNER;

  // xz = x @ in_w^T (M, 2048), fp32 (feeds conv + z gate)
  launch_gemm(xinb, D_MODEL, in_w, D_MODEL, nullptr, xz, nullptr, 2 * D_INNER,
              M, 2 * D_INNER, D_MODEL, 0, s);
  // xc = silu(depthwise causal conv + bias), fp32 + bf16
  conv_silu_kernel<<<(M * D_INNER + 255) / 256, 256, 0, s>>>(
      xz, xc, xcb, conv_w + (size_t)layer * D_INNER * D_CONV,
      conv_b + (size_t)layer * D_INNER, M, L);
  // dbl = xc @ xproj^T (M, 64) = [dt(32)|B(16)|C(16)], fp32 (scan) + bf16 (dt gemm)
  launch_gemm(xcb, D_INNER, xproj, D_INNER, nullptr, dbl, dblb, 64,
              M, DT_RANK + 2 * D_STATE, D_INNER, 0, s);
  // dt = softplus(dbl[:, :32] @ dtw^T + dtb) (M, 1024), fp32
  launch_gemm(dblb, 64, dtw, DT_RANK, dtb + (size_t)layer * D_INNER,
              dt, nullptr, D_INNER, M, D_INNER, DT_RANK, 2, s);
  // y = scan fused with +xc*D and *silu(z) -> bf16
  scan_kernel<<<(BATCH * D_INNER) / 256, 256, 0, s>>>(dbl, dt, xc, xz + D_INNER,
      Alog + (size_t)layer * D_INNER * D_STATE, Dsk + (size_t)layer * D_INNER,
      ysb, L);
  // t2 = y @ outw^T (M, 512), fp32 (residual)
  launch_gemm(ysb, D_INNER, outw, D_INNER, nullptr, t2, nullptr, D_MODEL,
              M, D_MODEL, D_INNER, 0, s);
}

extern "C" void kernel_launch(void* const* d_in, const int* in_sizes, int n_in,
                              void* d_out, int out_size, void* d_ws, size_t ws_size,
                              hipStream_t stream)
{
  (void)in_sizes; (void)n_in; (void)out_size; (void)ws_size;
  const float* src     = (const float*)d_in[0];
  const int*   tgt     = (const int*)  d_in[1];
  const float* emb     = (const float*)d_in[2];
  const float* in_w_s  = (const float*)d_in[3];
  const float* conv_w_s= (const float*)d_in[4];
  const float* conv_b_s= (const float*)d_in[5];
  const float* xproj_s = (const float*)d_in[6];
  const float* dtw_s   = (const float*)d_in[7];
  const float* dtb_s   = (const float*)d_in[8];
  const float* Alog_s  = (const float*)d_in[9];
  const float* D_s     = (const float*)d_in[10];
  const float* outw_s  = (const float*)d_in[11];
  const float* in_w_c  = (const float*)d_in[12];
  const float* conv_w_c= (const float*)d_in[13];
  const float* conv_b_c= (const float*)d_in[14];
  const float* xproj_c = (const float*)d_in[15];
  const float* dtw_c   = (const float*)d_in[16];
  const float* dtb_c   = (const float*)d_in[17];
  const float* Alog_c  = (const float*)d_in[18];
  const float* D_c     = (const float*)d_in[19];
  const float* outw_c  = (const float*)d_in[20];
  const float* ffn_w1  = (const float*)d_in[21];
  const float* ffn_b1  = (const float*)d_in[22];
  const float* ffn_w2  = (const float*)d_in[23];
  const float* ffn_b2  = (const float*)d_in[24];
  const float* ln_g    = (const float*)d_in[25];
  const float* ln_b    = (const float*)d_in[26];
  const float* fin_g   = (const float*)d_in[27];
  const float* fin_b   = (const float*)d_in[28];

  const int LC = LS + LT;                             // 1024 (cross length)
  const size_t nInW  = (size_t)N_LAYERS * 2 * D_INNER * D_MODEL;  // 6.29M
  const size_t nXprj = (size_t)N_LAYERS * 64 * D_INNER;
  const size_t nDtw  = (size_t)N_LAYERS * D_INNER * DT_RANK;
  const size_t nOutW = (size_t)N_LAYERS * D_MODEL * D_INNER;
  const size_t nFfn  = (size_t)N_LAYERS * D_FFN * D_MODEL;

  // ---- fp32 workspace region
  float* w = (float*)d_ws;
  float* X   = w; w += (size_t)BATCH * LT * D_MODEL;        // (2,512,512)
  float* XZ  = w; w += (size_t)BATCH * LC * 2 * D_INNER;    // (2,1024,2048)
  float* XC  = w; w += (size_t)BATCH * LC * D_INNER;        // (2,1024,1024)
  float* DBL = w; w += (size_t)BATCH * LC * 64;             // (2,1024,64)
  float* DT  = w; w += (size_t)BATCH * LC * D_INNER;        // (2,1024,1024)
  float* T2  = w; w += (size_t)BATCH * LC * D_MODEL;        // (2,1024,512)
  // ---- bf16 workspace region
  __bf16* wb = (__bf16*)w;
  __bf16* Xb   = wb; wb += (size_t)BATCH * LT * D_MODEL;
  __bf16* CATb = wb; wb += (size_t)BATCH * LC * D_MODEL;
  __bf16* XCb  = wb; wb += (size_t)BATCH * LC * D_INNER;
  __bf16* DBLb = wb; wb += (size_t)BATCH * LC * 64;
  __bf16* YSb  = wb; wb += (size_t)BATCH * LC * D_INNER;
  __bf16* HFb  = wb; wb += (size_t)BATCH * LT * D_FFN;
  MambaW Ws, Wc;
  Ws.in_w  = wb; wb += nInW;
  Ws.xproj = wb; wb += nXprj;
  Ws.dtw   = wb; wb += nDtw;
  Ws.outw  = wb; wb += nOutW;
  Wc.in_w  = wb; wb += nInW;
  Wc.xproj = wb; wb += nXprj;
  Wc.dtw   = wb; wb += nDtw;
  Wc.outw  = wb; wb += nOutW;
  __bf16* FW1b = wb; wb += nFfn;
  __bf16* FW2b = wb; wb += nFfn;

  // ---- pack all weights to bf16 once per launch
  launch_pack(in_w_s,  Ws.in_w,  nInW,  stream);
  launch_pack(xproj_s, Ws.xproj, nXprj, stream);
  launch_pack(dtw_s,   Ws.dtw,   nDtw,  stream);
  launch_pack(outw_s,  Ws.outw,  nOutW, stream);
  launch_pack(in_w_c,  Wc.in_w,  nInW,  stream);
  launch_pack(xproj_c, Wc.xproj, nXprj, stream);
  launch_pack(dtw_c,   Wc.dtw,   nDtw,  stream);
  launch_pack(outw_c,  Wc.outw,  nOutW, stream);
  launch_pack(ffn_w1,  FW1b,     nFfn,  stream);
  launch_pack(ffn_w2,  FW2b,     nFfn,  stream);

  embed_kernel<<<(BATCH * LT * D_MODEL + 255) / 256, 256, 0, stream>>>(tgt, emb, X, Xb);

  for (int i = 0; i < N_LAYERS; ++i) {
    // --- self mamba + post-norm
    run_mamba(Xb, LT, i, Ws, conv_w_s, conv_b_s, dtb_s, Alog_s, D_s,
              XZ, XC, XCb, DBL, DBLb, DT, YSb, T2, stream);
    add_ln_kernel<<<BATCH * LT, 256, 0, stream>>>(X, T2, LT, 0,
        ln_g + ((size_t)i * 3 + 0) * D_MODEL, ln_b + ((size_t)i * 3 + 0) * D_MODEL,
        X, Xb);

    // --- cross mamba over concat([src, x]) + post-norm (keep last LT rows)
    concat_kernel<<<(BATCH * LC * D_MODEL + 255) / 256, 256, 0, stream>>>(src, X, CATb);
    run_mamba(CATb, LC, i, Wc, conv_w_c, conv_b_c, dtb_c, Alog_c, D_c,
              XZ, XC, XCb, DBL, DBLb, DT, YSb, T2, stream);
    add_ln_kernel<<<BATCH * LT, 256, 0, stream>>>(X, T2, LC, LS,
        ln_g + ((size_t)i * 3 + 1) * D_MODEL, ln_b + ((size_t)i * 3 + 1) * D_MODEL,
        X, Xb);

    // --- FFN + post-norm
    launch_gemm(Xb, D_MODEL, FW1b + (size_t)i * D_FFN * D_MODEL, D_MODEL,
                ffn_b1 + (size_t)i * D_FFN, nullptr, HFb, D_FFN,
                BATCH * LT, D_FFN, D_MODEL, 1, stream);
    launch_gemm(HFb, D_FFN, FW2b + (size_t)i * D_MODEL * D_FFN, D_FFN,
                ffn_b2 + (size_t)i * D_MODEL, T2, nullptr, D_MODEL,
                BATCH * LT, D_MODEL, D_FFN, 0, stream);
    add_ln_kernel<<<BATCH * LT, 256, 0, stream>>>(X, T2, LT, 0,
        ln_g + ((size_t)i * 3 + 2) * D_MODEL, ln_b + ((size_t)i * 3 + 2) * D_MODEL,
        X, Xb);
  }

  // final LayerNorm -> d_out (fp32)
  add_ln_kernel<<<BATCH * LT, 256, 0, stream>>>(X, nullptr, LT, 0,
                                                fin_g, fin_b, (float*)d_out, nullptr);
}
